// GnoPremEq_26568667693797
// MI455X (gfx1250) — matrix operand order repaired and checked by reading, not compiled, not run on hardware
//
#include <hip/hip_runtime.h>
#include <hip/hip_bf16.h>

// ---------------- problem constants (match reference) ----------------
#define VAR_NUM 3
#define IN_DIM 8
#define OUT_DIM 32
#define N_PTS 8192
#define M_PTS 2048
#define K_LAST 10
#define PROJ_HIDDEN 64

#define WAVES 8
#define TPB (WAVES * 32)
#define LDW 96   // padded LDS row width in halfs (covers K up to 96 = 3 chunks of 32)

typedef __attribute__((ext_vector_type(16))) _Float16 v16h;
typedef __attribute__((ext_vector_type(8)))  _Float16 v8h;
typedef __attribute__((ext_vector_type(8)))  float    v8f;

// Branch-free erf (Abramowitz-Stegun 7.1.26, |err| < 1.5e-7) -> branch-free GELU.
// Division replaced by hardware v_rcp_f32 (~1 ulp; far below the f16 activation
// quantization noise of this WMMA pipeline).
__device__ __forceinline__ float erf_approx(float x) {
    const float ax = fabsf(x);
    const float t = __builtin_amdgcn_rcpf(1.0f + 0.3275911f * ax);
    float y = 1.061405429f;
    y = y * t - 1.453152027f;
    y = y * t + 1.421413741f;
    y = y * t - 0.284496736f;
    y = y * t + 0.254829592f;
    y = y * t;
    const float r = 1.0f - y * __expf(-ax * ax);
    return copysignf(r, x);
}
__device__ __forceinline__ float gelu_f(float x) {
    return 0.5f * x * (1.0f + erf_approx(x * 0.7071067811865476f));
}

// A fragment (16x32 f16) from LDS tile (16 rows x LDW halfs, zero padded).
// lane l holds row l&15; element i -> k = kBase + 16*(l>>4) + i
// => one contiguous 16-half run: [kBase+16g, +16)
__device__ __forceinline__ v16h frag_a(const _Float16* buf, int lane, int kBase) {
    const int m = lane & 15, g = lane >> 4;
    const _Float16* p = buf + m * LDW + kBase + 16 * g;
    const v8h lo = *(const v8h*)p;
    const v8h hi = *(const v8h*)(p + 8);
    v16h a;
#pragma unroll
    for (int i = 0; i < 8; ++i) { a[i] = lo[i]; a[i + 8] = hi[i]; }
    return a;
}

// B fragment (32x16 f16) from prepped weights Wt[N][KPAD] (f16, transposed,
// K zero-padded to a multiple of 32).
// lane l holds column nBase+(l&15); element i -> k = kBase + 16*(l>>4) + i
// => one contiguous 16-half (32B) run.
__device__ __forceinline__ v16h frag_b(const _Float16* __restrict__ Wt, int KPAD,
                                       int lane, int kBase, int nBase) {
    const int n = nBase + (lane & 15), g = lane >> 4;
    const _Float16* p = Wt + n * KPAD + kBase + 16 * g;
    const v8h lo = *(const v8h*)p;
    const v8h hi = *(const v8h*)(p + 8);
    v16h b;
#pragma unroll
    for (int i = 0; i < 8; ++i) { b[i] = lo[i]; b[i + 8] = hi[i]; }
    return b;
}

// One dense layer on a 16-row tile: in(16 x LDW f16) -> out(16 x LDW f16, zero padded).
// N must be a multiple of 16.  D layout: lane l, elem r -> m = 8*(l>>4)+r, n = l&15.
__device__ __forceinline__ void layer_f16(const _Float16* in, _Float16* out,
                                          const _Float16* __restrict__ Wt, int KPAD,
                                          const float* __restrict__ B,
                                          int N, bool act, int lane) {
    const int kchunks = KPAD >> 5;
    const int nl = lane & 15, g = lane >> 4;
    {   // vectorized zero-fill (padding columns for the next layer's K chunks)
        const v8h z = {};
        v8h* o8 = (v8h*)out;
        for (int i = lane; i < (16 * LDW) / 8; i += 32) o8[i] = z;
    }
    for (int nt = 0; nt < (N >> 4); ++nt) {
        const float bias = B[nt * 16 + nl];
        v8f c;
#pragma unroll
        for (int r = 0; r < 8; ++r) c[r] = bias;
        for (int kc = 0; kc < kchunks; ++kc) {
            v16h a = frag_a(in, lane, kc * 32);
            v16h b = frag_b(Wt, KPAD, lane, kc * 32, nt * 16);
            c = __builtin_amdgcn_wmma_f32_16x16x32_f16(false, a, false, b,
                                                       (short)0, c, false, false);
        }
#pragma unroll
        for (int r = 0; r < 8; ++r) {
            float v = c[r];
            if (act) v = gelu_f(v);
            out[(8 * g + r) * LDW + nt * 16 + nl] = (_Float16)v;
        }
    }
}

// Final layer, N == 32, f32 output tile (16 x 32 floats) written to LDS.
__device__ __forceinline__ void layer_f32out(const _Float16* in, float* out,
                                             const _Float16* __restrict__ Wt, int KPAD,
                                             const float* __restrict__ B, int lane) {
    const int kchunks = KPAD >> 5;
    const int nl = lane & 15, g = lane >> 4;
    for (int nt = 0; nt < 2; ++nt) {
        const float bias = B[nt * 16 + nl];
        v8f c;
#pragma unroll
        for (int r = 0; r < 8; ++r) c[r] = bias;
        for (int kc = 0; kc < kchunks; ++kc) {
            v16h a = frag_a(in, lane, kc * 32);
            v16h b = frag_b(Wt, KPAD, lane, kc * 32, nt * 16);
            c = __builtin_amdgcn_wmma_f32_16x16x32_f16(false, a, false, b,
                                                       (short)0, c, false, false);
        }
#pragma unroll
        for (int r = 0; r < 8; ++r)
            out[(8 * g + r) * 32 + nt * 16 + nl] = c[r];
    }
}

// ---------------- weight prep: W[K][N] f32 -> Wt[N][KPAD] f16 (transposed) ----
__global__ void prep_w_kernel(const float* __restrict__ W, _Float16* __restrict__ Wt,
                              int K, int N, int KPAD) {
    const int t = blockIdx.x * blockDim.x + threadIdx.x;
    if (t >= N * KPAD) return;
    const int n = t / KPAD, k = t - n * KPAD;
    Wt[t] = (k < K) ? (_Float16)W[k * N + n] : (_Float16)0.0f;
}

// ---------------- kernel 1: projection MLP (8 -> 64 gelu -> 32) ----------------
// rows = N_PTS * VAR_NUM, row = p*3 + v; inp flat is [p][v][8]; out proj[v][p][32]
__global__ void __launch_bounds__(TPB)
proj_kernel(const float* __restrict__ inp,
            const _Float16* __restrict__ W0t, const float* __restrict__ B0,
            const _Float16* __restrict__ W1t, const float* __restrict__ B1,
            float* __restrict__ proj, int rows) {
    __shared__ __align__(16) _Float16 smem[WAVES][2][16 * LDW];
    const int lane = threadIdx.x & 31;
    const int wv = threadIdx.x >> 5;
    const int e0 = (blockIdx.x * WAVES + wv) * 16;
    _Float16* bufA = smem[wv][0];
    _Float16* bufB = smem[wv][1];

    for (int idx = lane; idx < 16 * LDW; idx += 32) {
        const int r = idx / LDW, c = idx - r * LDW;
        const int row = e0 + r;
        float val = 0.0f;
        if (row < rows && c < IN_DIM) val = inp[row * IN_DIM + c];
        bufA[idx] = (_Float16)val;
    }
    __syncthreads();
    layer_f16(bufA, bufB, W0t, 32, B0, PROJ_HIDDEN, true, lane);
    __syncthreads();
    float* fout = (float*)bufA;
    layer_f32out(bufB, fout, W1t, 64, B1, lane);
    __syncthreads();
    for (int idx = lane; idx < 16 * 32; idx += 32) {
        const int r = idx >> 5, c = idx & 31;
        const int row = e0 + r;
        if (row < rows) {
            const int p = row / VAR_NUM, v = row - p * VAR_NUM;
            proj[(v * N_PTS + p) * 32 + c] = fout[idx];
        }
    }
}

// ---------------- kernel 2: edge MLP (36 -> 80 -> 80 -> 32) + segment atomics ----
__global__ void __launch_bounds__(TPB)
edge_mlp_kernel(const float* __restrict__ input_grid,
                const float* __restrict__ proj,        // [3][N_PTS][32]
                const int* __restrict__ nbr_index,
                const int* __restrict__ nbr_seg,
                const _Float16* __restrict__ W0t, const float* __restrict__ B0,
                const _Float16* __restrict__ W1t, const float* __restrict__ B1,
                const _Float16* __restrict__ W2t, const float* __restrict__ B2,
                float* __restrict__ fsum,              // [3][N_PTS][32]
                int E) {
    __shared__ __align__(16) _Float16 smem[WAVES][2][16 * LDW];
    const int lane = threadIdx.x & 31;
    const int wv = threadIdx.x >> 5;
    const int var = blockIdx.y;
    const int e0 = (blockIdx.x * WAVES + wv) * 16;
    _Float16* bufA = smem[wv][0];
    _Float16* bufB = smem[wv][1];

    for (int idx = lane; idx < 16 * LDW; idx += 32) {
        const int r = idx / LDW, c = idx - r * LDW;
        const int e = e0 + r;
        float val = 0.0f;
        if (e < E && c < 36) {
            if (c < 2)       val = input_grid[nbr_index[e] * 2 + c];
            else if (c < 4)  val = input_grid[nbr_seg[e] * 2 + (c - 2)];
            else             val = proj[(var * N_PTS + nbr_index[e]) * 32 + (c - 4)];
        }
        bufA[idx] = (_Float16)val;
    }
    __syncthreads();
    layer_f16(bufA, bufB, W0t, 64, B0, 80, true, lane);
    __syncthreads();
    layer_f16(bufB, bufA, W1t, 96, B1, 80, true, lane);
    __syncthreads();
    float* fout = (float*)bufB;
    layer_f32out(bufA, fout, W2t, 96, B2, lane);
    __syncthreads();
    for (int idx = lane; idx < 16 * 32; idx += 32) {
        const int r = idx >> 5, c = idx & 31;
        const int e = e0 + r;
        if (e < E)
            atomicAdd(&fsum[(var * N_PTS + nbr_seg[e]) * 32 + c], fout[idx]);
    }
}

// ---------------- kernel 3: residual + mean combine --------------------------
__global__ void combine_kernel(const float* __restrict__ fsum,
                               const float* __restrict__ proj,
                               const int* __restrict__ counts,
                               float* __restrict__ fnew, int total) {
    const int t = blockIdx.x * blockDim.x + threadIdx.x;
    if (t >= total) return;
    const int i = (t >> 5) & (N_PTS - 1);
    const int cnt = counts[i];
    const float ic = 1.0f / (float)(cnt > 1 ? cnt : 1);
    fnew[t] = fsum[t] * ic + proj[t];
}

// ---------------- kernel 4: kNN MLP (36 -> 80 -> 80 -> 32), mean over K_LAST --
__global__ void __launch_bounds__(TPB)
last_mlp_kernel(const float* __restrict__ input_grid,
                const float* __restrict__ output_grid,
                const float* __restrict__ fnew,        // [3][N_PTS][32]
                const int* __restrict__ nbr_last,      // [M_PTS*K_LAST]
                const _Float16* __restrict__ W0t, const float* __restrict__ B0,
                const _Float16* __restrict__ W1t, const float* __restrict__ B1,
                const _Float16* __restrict__ W2t, const float* __restrict__ B2,
                float* __restrict__ out,               // [M_PTS][96]
                int rows) {
    __shared__ __align__(16) _Float16 smem[WAVES][2][16 * LDW];
    const int lane = threadIdx.x & 31;
    const int wv = threadIdx.x >> 5;
    const int var = blockIdx.y;
    const int e0 = (blockIdx.x * WAVES + wv) * 16;
    _Float16* bufA = smem[wv][0];
    _Float16* bufB = smem[wv][1];

    for (int idx = lane; idx < 16 * LDW; idx += 32) {
        const int r = idx / LDW, c = idx - r * LDW;
        const int q = e0 + r;
        float val = 0.0f;
        if (q < rows && c < 36) {
            if (c < 2)       val = input_grid[nbr_last[q] * 2 + c];
            else if (c < 4)  val = output_grid[(q / K_LAST) * 2 + (c - 2)];
            else             val = fnew[(var * N_PTS + nbr_last[q]) * 32 + (c - 4)];
        }
        bufA[idx] = (_Float16)val;
    }
    __syncthreads();
    layer_f16(bufA, bufB, W0t, 64, B0, 80, true, lane);
    __syncthreads();
    layer_f16(bufB, bufA, W1t, 96, B1, 80, true, lane);
    __syncthreads();
    float* fout = (float*)bufB;
    layer_f32out(bufA, fout, W2t, 96, B2, lane);
    __syncthreads();
    const float inv_k = 1.0f / (float)K_LAST;
    for (int idx = lane; idx < 16 * 32; idx += 32) {
        const int r = idx >> 5, c = idx & 31;
        const int q = e0 + r;
        if (q < rows)
            atomicAdd(&out[(q / K_LAST) * (VAR_NUM * OUT_DIM) + var * 32 + c],
                      fout[idx] * inv_k);
    }
}

// ---------------- host side ---------------------------------------------------
extern "C" void kernel_launch(void* const* d_in, const int* in_sizes, int n_in,
                              void* d_out, int out_size, void* d_ws, size_t ws_size,
                              hipStream_t stream) {
    const float* inp         = (const float*)d_in[0];
    const float* input_grid  = (const float*)d_in[1];
    const float* output_grid = (const float*)d_in[2];
    const float* pW0 = (const float*)d_in[3];
    const float* pb0 = (const float*)d_in[4];
    const float* pW1 = (const float*)d_in[5];
    const float* pb1 = (const float*)d_in[6];
    const float* i0W0 = (const float*)d_in[7];
    const float* i0b0 = (const float*)d_in[8];
    const float* i0W1 = (const float*)d_in[9];
    const float* i0b1 = (const float*)d_in[10];
    const float* i0W2 = (const float*)d_in[11];
    const float* i0b2 = (const float*)d_in[12];
    const float* i1W0 = (const float*)d_in[13];
    const float* i1b0 = (const float*)d_in[14];
    const float* i1W1 = (const float*)d_in[15];
    const float* i1b1 = (const float*)d_in[16];
    const float* i1W2 = (const float*)d_in[17];
    const float* i1b2 = (const float*)d_in[18];
    const int* nbr_index  = (const int*)d_in[19];
    const int* nbr_seg    = (const int*)d_in[20];
    const int* nbr_counts = (const int*)d_in[21];
    const int* nbr_last   = (const int*)d_in[22];
    const int E = in_sizes[19];

    const int per = VAR_NUM * N_PTS * 32;   // 786432 floats per buffer
    float* proj = (float*)d_ws;
    float* fsum = proj + per;
    float* fnew = fsum + per;

    // prepped f16 transposed weights live after the f32 scratch
    _Float16* wb = (_Float16*)(fnew + per);
    _Float16* pW0t  = wb;                   // N=64, KPAD=32  -> 2048
    _Float16* pW1t  = pW0t + 64 * 32;       // N=32, KPAD=64  -> 2048
    _Float16* e0W0t = pW1t + 32 * 64;       // N=80, KPAD=64  -> 5120
    _Float16* e0W1t = e0W0t + 80 * 64;      // N=80, KPAD=96  -> 7680
    _Float16* e0W2t = e0W1t + 80 * 96;      // N=32, KPAD=96  -> 3072
    _Float16* e1W0t = e0W2t + 32 * 96;      // N=80, KPAD=64  -> 5120
    _Float16* e1W1t = e1W0t + 80 * 64;      // N=80, KPAD=96  -> 7680
    _Float16* e1W2t = e1W1t + 80 * 96;      // N=32, KPAD=96  -> 3072

    hipMemsetAsync(fsum, 0, (size_t)per * sizeof(float), stream);
    hipMemsetAsync(d_out, 0, (size_t)out_size * sizeof(float), stream);

    // ---- prep weights (tiny) ----
    struct { const float* W; _Float16* Wt; int K, N, KPAD; } prep[8] = {
        { pW0,  pW0t,   8, 64, 32 },
        { pW1,  pW1t,  64, 32, 64 },
        { i0W0, e0W0t, 36, 80, 64 },
        { i0W1, e0W1t, 80, 80, 96 },
        { i0W2, e0W2t, 80, 32, 96 },
        { i1W0, e1W0t, 36, 80, 64 },
        { i1W1, e1W1t, 80, 80, 96 },
        { i1W2, e1W2t, 80, 32, 96 },
    };
    for (int i = 0; i < 8; ++i) {
        const int tot = prep[i].N * prep[i].KPAD;
        prep_w_kernel<<<(tot + 255) / 256, 256, 0, stream>>>(
            prep[i].W, prep[i].Wt, prep[i].K, prep[i].N, prep[i].KPAD);
    }

    {   // projection
        const int rows = N_PTS * VAR_NUM;
        const int tiles = (rows + 15) / 16;
        int blocks = (tiles + WAVES - 1) / WAVES;
        if (blocks < 1) blocks = 1;
        proj_kernel<<<blocks, TPB, 0, stream>>>(inp, pW0t, pb0, pW1t, pb1, proj, rows);
    }
    {   // edge MLP, all 3 vars
        const int tiles = (E + 15) / 16;
        int bx = (tiles + WAVES - 1) / WAVES;
        if (bx < 1) bx = 1;
        dim3 grid(bx, VAR_NUM);
        edge_mlp_kernel<<<grid, TPB, 0, stream>>>(input_grid, proj, nbr_index, nbr_seg,
                                                  e0W0t, i0b0, e0W1t, i0b1, e0W2t, i0b2,
                                                  fsum, E);
    }
    {   // mean + residual
        const int total = per;
        combine_kernel<<<(total + 255) / 256, 256, 0, stream>>>(fsum, proj, nbr_counts,
                                                                fnew, total);
    }
    {   // kNN MLP + mean over K_LAST
        const int rows = M_PTS * K_LAST;
        const int tiles = (rows + 15) / 16;
        int bx = (tiles + WAVES - 1) / WAVES;
        if (bx < 1) bx = 1;
        dim3 grid(bx, VAR_NUM);
        last_mlp_kernel<<<grid, TPB, 0, stream>>>(input_grid, output_grid, fnew, nbr_last,
                                                  e1W0t, i1b0, e1W1t, i1b1, e1W2t, i1b2,
                                                  (float*)d_out, rows);
    }
}
